// CausalNeuralHawkesMasked_36988258353421
// MI455X (gfx1250) — compile-verified, matrix operand order checked by reading, not run on hardware
//
#include <hip/hip_runtime.h>
#include <hip/hip_bf16.h>

#define B_DIM   512
#define S_DIM   2048
#define T_STEPS 2047
#define HID     32
#define NT      14      // 224 / 16 column tiles
#define EMB_ROWS 21
#define EMB_STR 40      // halfs, 80B row stride (16B aligned)
#define H_STR   40      // halfs
#define W_STR   72      // halfs, 144B row stride (16B aligned)

typedef __bf16 bf16;
typedef bf16  v8bf  __attribute__((ext_vector_type(8)));
typedef bf16  v16bf __attribute__((ext_vector_type(16)));
typedef float v8f   __attribute__((ext_vector_type(8)));

// Assemble one wave-lane's 16 bf16 A/B elements from two 16-byte LDS chunks
// (ISA 16-bit A layout: VGPR0-3 hold K = 8*hk .. 8*hk+7, VGPR4-7 hold K = 16+8*hk ..).
__device__ __forceinline__ v16bf ld_tile(const bf16* p0, const bf16* p1) {
    v8bf a = *(const v8bf*)p0;
    v8bf b = *(const v8bf*)p1;
    v16bf r;
#pragma unroll
    for (int i = 0; i < 8; ++i) { r[i] = a[i]; r[i + 8] = b[i]; }
    return r;
}

__device__ __forceinline__ float sigm(float x)   { return 1.0f / (1.0f + __expf(-x)); }
__device__ __forceinline__ float tanh_f(float x) { float e = __expf(2.0f * x); return 1.0f - 2.0f / (e + 1.0f); }

__global__ __launch_bounds__(32)
void hawkes_ctlstm_kernel(const int*   __restrict__ types,
                          const float* __restrict__ dtime,
                          const float* __restrict__ emb,
                          const float* __restrict__ W,
                          const float* __restrict__ b,
                          float*       __restrict__ out)
{
    __shared__ __align__(16) bf16  s_emb[EMB_ROWS * EMB_STR];
    __shared__ __align__(16) bf16  s_W[224 * W_STR];     // W transposed: s_W[n*W_STR + k], bf16
    __shared__ __align__(16) bf16  s_h[16 * H_STR];      // h carry, bf16, row = batch row in tile
    __shared__ float s_dt[16];

    const int lane = threadIdx.x;     // 0..31, one wave per block
    const int n    = lane & 15;       // column / A-row within tile
    const int hk   = lane >> 4;       // half-wave select
    const int base = blockIdx.x * 16; // batch-row tile base

    // ---- one-time preload (single wave => no barriers; DS is in-order per wave) ----
    for (int i = lane; i < EMB_ROWS * HID; i += 32) {
        int r = i >> 5, c = i & 31;
        s_emb[r * EMB_STR + c] = (bf16)emb[i];
    }
    for (int i = lane; i < 64 * 224; i += 32) {        // transpose W (64x224) -> Wt (224x64)
        int k = i / 224, nn = i - k * 224;
        s_W[nn * W_STR + k] = (bf16)W[i];
    }
    for (int i = lane; i < 16 * H_STR; i += 32) s_h[i] = (bf16)0.0f;

    float bias[NT];
#pragma unroll
    for (int nt = 0; nt < NT; ++nt) bias[nt] = b[16 * nt + n];

    float c_st[16], cb_st[16];
#pragma unroll
    for (int i = 0; i < 16; ++i) { c_st[i] = 0.0f; cb_st[i] = 0.0f; }

    const size_t TBH = (size_t)T_STEPS * B_DIM * HID;

    for (int t = 0; t < T_STEPS; ++t) {
        // per-lane scalar fetches: event type for batch row (base+n), dt for decay
        int   typ = types[(size_t)(base + n) * S_DIM + t];
        float dtv = dtime[(size_t)(base + n) * S_DIM + t + 1];
        if (lane < 16) s_dt[lane] = dtv;   // row 'lane' of this tile

        // A matrices: x chunk (K 0..31) from emb row, h chunk (K 32..63) from carry
        const bf16* er = s_emb + typ * EMB_STR;
        v16bf ax = ld_tile(er + 8 * hk, er + 16 + 8 * hk);
        const bf16* hr = s_h + n * H_STR;
        v16bf ah = ld_tile(hr + 8 * hk, hr + 16 + 8 * hk);

        // g = [x,h] @ W + b : 14 N-tiles x 2 bf16 WMMAs (K=64 total)
        v8f acc[NT];
#pragma unroll
        for (int nt = 0; nt < NT; ++nt) {
            const bf16* wr = s_W + (16 * nt + n) * W_STR;
            v16bf b0 = ld_tile(wr + 8 * hk,      wr + 16 + 8 * hk);   // K 0..31
            v16bf b1 = ld_tile(wr + 32 + 8 * hk, wr + 48 + 8 * hk);   // K 32..63
            v8f c;
#pragma unroll
            for (int i = 0; i < 8; ++i) c[i] = bias[nt];
            c = __builtin_amdgcn_wmma_f32_16x16x32_bf16(false, ax, false, b0, (short)0, c, false, false);
            c = __builtin_amdgcn_wmma_f32_16x16x32_bf16(false, ah, false, b1, (short)0, c, false, false);
            acc[nt] = c;
        }

        // dt for the 8 rows this lane-half owns (rows 8*hk .. 8*hk+7)
        float dts[8];
#pragma unroll
        for (int v = 0; v < 8; ++v) dts[v] = s_dt[8 * hk + v];

        // C/D layout: component v -> row (v + 8*hk); lane n -> hid (16*tt + n).
        // All 7 gates of a given element live in the same lane/slot across tiles.
        size_t obase = (size_t)t * B_DIM * HID + (size_t)(base + 8 * hk) * HID + n;

#pragma unroll
        for (int tt = 0; tt < 2; ++tt) {
#pragma unroll
            for (int v = 0; v < 8; ++v) {
                float gi  = acc[0 + tt][v];
                float gf  = acc[2 + tt][v];
                float gz  = acc[4 + tt][v];
                float go  = acc[6 + tt][v];
                float gib = acc[8 + tt][v];
                float gfb = acc[10 + tt][v];
                float gd  = acc[12 + tt][v];

                float ig  = sigm(gi);
                float fg  = sigm(gf);
                float zg  = tanh_f(gz);
                float og  = sigm(go);
                float ibg = sigm(gib);
                float fbg = sigm(gfb);
                float delta = 10.0f * __logf(1.0f + __expf(0.1f * gd));  // softplus(beta*x)/beta

                int si = tt * 8 + v;
                float cn  = fg * c_st[si] + ig * zg;
                float cbn = fbg * cb_st[si] + ibg * zg;
                float ct  = cbn + (cn - cbn) * __expf(-delta * dts[v]);
                float hn  = og * tanh_f(ct);

                c_st[si]  = ct;    // carry is decayed cell
                cb_st[si] = cbn;
                s_h[(v + 8 * hk) * H_STR + 16 * tt + n] = (bf16)hn;  // carry for next step's A

                size_t e = obase + (size_t)v * HID + 16 * tt;
                __builtin_nontemporal_store(hn,    &out[e]);            // h_out
                __builtin_nontemporal_store(cn,    &out[TBH + e]);      // c_out (pre-decay)
                __builtin_nontemporal_store(cbn,   &out[2 * TBH + e]);  // c_bar_out
                __builtin_nontemporal_store(delta, &out[3 * TBH + e]);  // decay_out
                __builtin_nontemporal_store(og,    &out[4 * TBH + e]);  // gate_out
            }
        }
    }
}

extern "C" void kernel_launch(void* const* d_in, const int* in_sizes, int n_in,
                              void* d_out, int out_size, void* d_ws, size_t ws_size,
                              hipStream_t stream) {
    const int*   types = (const int*)  d_in[0];
    const float* dtime = (const float*)d_in[1];
    const float* emb   = (const float*)d_in[2];
    const float* W     = (const float*)d_in[3];
    const float* b     = (const float*)d_in[4];
    float* out = (float*)d_out;

    hawkes_ctlstm_kernel<<<dim3(B_DIM / 16), dim3(32), 0, stream>>>(types, dtime, emb, W, b, out);
}
